// GINModel_38749194945060
// MI455X (gfx1250) — matrix operand order repaired and checked by reading, not compiled, not run on hardware
//
#include <hip/hip_runtime.h>

typedef __attribute__((ext_vector_type(16))) _Float16 v16h;
typedef __attribute__((ext_vector_type(8)))  _Float16 h8;   // 16B
typedef __attribute__((ext_vector_type(4)))  _Float16 h4;   // 8B
typedef __attribute__((ext_vector_type(8)))  float    v8f;

union AV { v16h v; h8 p[2]; };

#define NN 100000
#define EE 1600000
#define GG 64

// ---------------------------------------------------------------- zero fill
__global__ void zero_kernel(float* __restrict__ p, int n) {
    int i = blockIdx.x * blockDim.x + threadIdx.x;
    if (i < n) p[i] = 0.0f;
}

// ------------------------------------------------- edge scatter-add (f32 atomics)
// agg[dst] += h[src], feature dim F, one thread per (edge, 4-float chunk)
template<int F>
__global__ void agg_kernel(const float* __restrict__ h,
                           const int* __restrict__ ei,
                           float* __restrict__ agg) {
    const int chunks = F / 4;
    long long tid = (long long)blockIdx.x * blockDim.x + threadIdx.x;
    if (tid >= (long long)EE * chunks) return;
    int e = (int)(tid / chunks);
    int c = (int)(tid % chunks) * 4;
    int src = ei[e];          // edge_index[0]
    int dst = ei[EE + e];     // edge_index[1]
    const float4 v = *(const float4*)(h + (size_t)src * F + c);
    float* p = agg + (size_t)dst * F + c;
    atomicAdd(p + 0, v.x);
    atomicAdd(p + 1, v.y);
    atomicAdd(p + 2, v.z);
    atomicAdd(p + 3, v.w);
}

// ------------------------------------------------- fused GIN MLP (WMMA f16->f32)
// out = relu( relu((h+agg) @ w1 + b1) @ w2 + b2 ),  DIN -> 64 -> 64
// 128 threads = 4 waves; each wave computes a 16-node tile.
template<int DIN>
__global__ void __launch_bounds__(128)
gin_mlp_kernel(const float* __restrict__ h,
               const float* __restrict__ agg,
               const float* __restrict__ w1, const float* __restrict__ b1,
               const float* __restrict__ w2, const float* __restrict__ b2,
               float* __restrict__ out) {
    constexpr int DINP = DIN + 8;   // padded K stride (halves) for bank spread
    constexpr int HP   = 72;        // padded 64

    __shared__ __align__(16) _Float16 wT1[64 * DINP];   // w1^T: [n][k]
    __shared__ __align__(16) _Float16 wT2[64 * HP];     // w2^T: [n][k]
    __shared__ __align__(16) _Float16 zbuf[4 * 16 * DINP];
    __shared__ __align__(16) _Float16 ybuf[4 * 16 * HP];
    __shared__ float b1s[64];
    __shared__ float b2s[64];

    // one-time cooperative weight transpose+convert into LDS
    for (int e = threadIdx.x; e < DIN * 64; e += 128) {
        int k = e >> 6, n = e & 63;
        wT1[n * DINP + k] = (_Float16)w1[e];
    }
    for (int e = threadIdx.x; e < 64 * 64; e += 128) {
        int k = e >> 6, n = e & 63;
        wT2[n * HP + k] = (_Float16)w2[e];
    }
    if (threadIdx.x < 64) {
        b1s[threadIdx.x] = b1[threadIdx.x];
        b2s[threadIdx.x] = b2[threadIdx.x];
    }
    __syncthreads();

    const int wave = threadIdx.x >> 5;
    const int lane = threadIdx.x & 31;
    const int node_base = (blockIdx.x * 4 + wave) * 16;
    if (node_base >= NN) return;          // wave-uniform (EXEC all-1s for WMMA)

    _Float16* zl = zbuf + wave * 16 * DINP;
    _Float16* yl = ybuf + wave * 16 * HP;

    // z = h + agg  ->  f16 LDS tile [16][DIN]   (fully coalesced float4 copy)
    {
        const float* hrow = h   + (size_t)node_base * DIN;
        const float* arow = agg + (size_t)node_base * DIN;
        for (int i = lane; i < (16 * DIN) / 4; i += 32) {
            int idx = i * 4;
            int m = idx / DIN;
            int c = idx % DIN;
            float4 hv = *(const float4*)(hrow + idx);
            float4 av = *(const float4*)(arow + idx);
            h4 zv;
            zv.x = (_Float16)(hv.x + av.x);
            zv.y = (_Float16)(hv.y + av.y);
            zv.z = (_Float16)(hv.z + av.z);
            zv.w = (_Float16)(hv.w + av.w);
            *(h4*)(zl + m * DINP + c) = zv;
        }
    }

    const int m    = lane & 15;
    const int ka   = (lane < 16) ? 0 : 8;    // A K-base (ISA 16-bit A layout)
    const int kb   = (lane < 16) ? 0 : 8;    // B K-base (same K order as A)
    const int mrow = (lane < 16) ? 0 : 8;    // C/D row base

    // ---- stage 1: y = relu(z @ w1 + b1)  (16 x 64) ----
    v8f acc[4] = {};
    for (int kc = 0; kc < DIN; kc += 32) {
        AV au;
        const _Float16* ap = zl + m * DINP + kc + ka;
        au.p[0] = *(const h8*)(ap);        // K = ka+0..7
        au.p[1] = *(const h8*)(ap + 16);   // K = ka+16..23
#pragma unroll
        for (int t = 0; t < 4; ++t) {
            AV bu;
            const _Float16* bp = wT1 + (t * 16 + m) * DINP + kc + kb;
            bu.p[0] = *(const h8*)(bp);        // K = kb+0..7
            bu.p[1] = *(const h8*)(bp + 16);   // K = kb+16..23
            acc[t] = __builtin_amdgcn_wmma_f32_16x16x32_f16(
                false, au.v, false, bu.v, (short)0, acc[t], false, false);
        }
    }
#pragma unroll
    for (int t = 0; t < 4; ++t) {
        int n = t * 16 + m;
        float bias = b1s[n];
#pragma unroll
        for (int r = 0; r < 8; ++r) {
            float v = fmaxf(acc[t][r] + bias, 0.0f);
            yl[(mrow + r) * HP + n] = (_Float16)v;   // C-layout -> [m][n] LDS
        }
    }

    // ---- stage 2: out = relu(y @ w2 + b2)  (16 x 64) ----
    v8f acc2[4] = {};
#pragma unroll
    for (int kc = 0; kc < 64; kc += 32) {
        AV au;
        const _Float16* ap = yl + m * HP + kc + ka;
        au.p[0] = *(const h8*)(ap);
        au.p[1] = *(const h8*)(ap + 16);
#pragma unroll
        for (int t = 0; t < 4; ++t) {
            AV bu;
            const _Float16* bp = wT2 + (t * 16 + m) * HP + kc + kb;
            bu.p[0] = *(const h8*)(bp);
            bu.p[1] = *(const h8*)(bp + 16);
            acc2[t] = __builtin_amdgcn_wmma_f32_16x16x32_f16(
                false, au.v, false, bu.v, (short)0, acc2[t], false, false);
        }
    }
    float* orow = out + (size_t)node_base * 64;
#pragma unroll
    for (int t = 0; t < 4; ++t) {
        int n = t * 16 + m;
        float bias = b2s[n];
#pragma unroll
        for (int r = 0; r < 8; ++r) {
            float v = fmaxf(acc2[t][r] + bias, 0.0f);
            orow[(size_t)(mrow + r) * 64 + n] = v;
        }
    }
}

// ------------------------------------------------- mean-pool accumulation
__global__ void pool_kernel(const float* __restrict__ h,
                            const int* __restrict__ batch,
                            float* __restrict__ sums,
                            float* __restrict__ cnt) {
    long long tid = (long long)blockIdx.x * blockDim.x + threadIdx.x;
    if (tid >= (long long)NN * 16) return;
    int node = (int)(tid >> 4);
    int c = (int)(tid & 15) * 4;
    int g = batch[node];
    float4 v = *(const float4*)(h + (size_t)node * 64 + c);
    atomicAdd(&sums[g * 64 + c + 0], v.x);
    atomicAdd(&sums[g * 64 + c + 1], v.y);
    atomicAdd(&sums[g * 64 + c + 2], v.z);
    atomicAdd(&sums[g * 64 + c + 3], v.w);
    if (c == 0) atomicAdd(&cnt[g], 1.0f);
}

// ------------------------------------------------- tiny classifier head
// out[g] = relu(pooled @ lin1 + b1) @ lin2 + b2,   pooled = sums[g]/max(cnt,1)
__global__ void head_kernel(const float* __restrict__ sums,
                            const float* __restrict__ cnt,
                            const float* __restrict__ w1, const float* __restrict__ b1,
                            const float* __restrict__ w2, const float* __restrict__ b2,
                            float* __restrict__ out) {
    int g = threadIdx.x;
    if (g >= GG) return;
    float c = fmaxf(cnt[g], 1.0f);
    float p[64];
    for (int j = 0; j < 64; ++j) p[j] = sums[g * 64 + j] / c;
    float y[64];
    for (int j = 0; j < 64; ++j) {
        float a = b1[j];
        for (int k = 0; k < 64; ++k) a += p[k] * w1[k * 64 + j];
        y[j] = fmaxf(a, 0.0f);
    }
    for (int cc = 0; cc < 2; ++cc) {
        float a = b2[cc];
        for (int k = 0; k < 64; ++k) a += y[k] * w2[k * 2 + cc];
        out[g * 2 + cc] = a;
    }
}

// ------------------------------------------------------------------ launch
extern "C" void kernel_launch(void* const* d_in, const int* in_sizes, int n_in,
                              void* d_out, int out_size, void* d_ws, size_t ws_size,
                              hipStream_t stream) {
    (void)in_sizes; (void)n_in; (void)out_size; (void)ws_size;

    const float* x     = (const float*)d_in[0];
    const int*   ei    = (const int*)d_in[1];
    const int*   batch = (const int*)d_in[2];
    const float* W[12];
    for (int i = 0; i < 12; ++i) W[i] = (const float*)d_in[3 + i];
    const float* lin1_w = (const float*)d_in[15];
    const float* lin1_b = (const float*)d_in[16];
    const float* lin2_w = (const float*)d_in[17];
    const float* lin2_b = (const float*)d_in[18];
    float* outp = (float*)d_out;

    // workspace layout (floats)
    float* agg  = (float*)d_ws;                       // N*128
    float* hA   = agg  + (size_t)NN * 128;            // N*64
    float* hB   = hA   + (size_t)NN * 64;             // N*64
    float* sums = hB   + (size_t)NN * 64;             // G*64
    float* cnt  = sums + GG * 64;                     // G   (contiguous w/ sums)

    auto zero = [&](float* p, long long n) {
        zero_kernel<<<(int)((n + 255) / 256), 256, 0, stream>>>(p, (int)n);
    };
    const int mlp_blocks = (NN / 16 + 3) / 4;         // 4 tiles (waves) per block

    // ---- layer 0 (DIN = 128) ----
    zero(agg, (long long)NN * 128);
    {
        long long tot = (long long)EE * 32;
        agg_kernel<128><<<(int)((tot + 255) / 256), 256, 0, stream>>>(x, ei, agg);
    }
    gin_mlp_kernel<128><<<mlp_blocks, 128, 0, stream>>>(x, agg, W[0], W[1], W[2], W[3], hA);

    // ---- layer 1 (DIN = 64) ----
    zero(agg, (long long)NN * 64);
    {
        long long tot = (long long)EE * 16;
        agg_kernel<64><<<(int)((tot + 255) / 256), 256, 0, stream>>>(hA, ei, agg);
    }
    gin_mlp_kernel<64><<<mlp_blocks, 128, 0, stream>>>(hA, agg, W[4], W[5], W[6], W[7], hB);

    // ---- layer 2 (DIN = 64) ----
    zero(agg, (long long)NN * 64);
    {
        long long tot = (long long)EE * 16;
        agg_kernel<64><<<(int)((tot + 255) / 256), 256, 0, stream>>>(hB, ei, agg);
    }
    gin_mlp_kernel<64><<<mlp_blocks, 128, 0, stream>>>(hB, agg, W[8], W[9], W[10], W[11], hA);

    // ---- mean pool + head ----
    zero(sums, GG * 64 + GG);
    {
        long long tot = (long long)NN * 16;
        pool_kernel<<<(int)((tot + 255) / 256), 256, 0, stream>>>(hA, batch, sums, cnt);
    }
    head_kernel<<<1, 64, 0, stream>>>(sums, cnt, lin1_w, lin1_b, lin2_w, lin2_b, outp);
}
